// AttentionLayer_8418135900302
// MI455X (gfx1250) — matrix operand order repaired and checked by reading, not compiled, not run on hardware
//
#include <hip/hip_runtime.h>
#include <math.h>

#define SEQ    4096
#define DM     512
#define NBATCH 8
#define LDSK   64   // K-width of a staged LDS tile (two K=32 WMMA sub-steps)
#define LDSS   72   // LDS row stride in halves (64 data + 4dw pad -> conflict-free)

typedef __attribute__((ext_vector_type(16))) _Float16 v16h;
typedef __attribute__((ext_vector_type(8)))  _Float16 v8h;
typedef __attribute__((ext_vector_type(8)))  float    v8f;

typedef __attribute__((ext_vector_type(4))) unsigned int tdm_g0_t;
typedef __attribute__((ext_vector_type(8))) int          tdm_g1_t;
typedef __attribute__((ext_vector_type(4))) int          tdm_g2_t;

// ---- TDM: DMA a 2D f16 tile (128 rows x LDSK halves) into LDS -------------
// LDS layout produced: row r at lds_addr + r*144B (32 data DWORDs + 4 pad
// DWORDs) == LDSS(=72) halves -> identical to the manual staging layout.
// Descriptor per CDNA5 ISA 8.3-8.6. Issue from ONE wave only (TDM ignores
// EXEC, so guard with a wave-uniform branch).
__device__ inline void tdm_load_2d(const _Float16* gptr, const _Float16* lptr,
                                   unsigned stride_halves) {
  unsigned long long ga = (unsigned long long)(const void*)gptr;
  unsigned lds = (unsigned)(unsigned long long)(const void*)lptr; // low32 = LDS offset
  tdm_g0_t g0;
  g0[0] = 1u;                                    // count=1 valid descriptor
  g0[1] = lds;                                   // lds_addr (bytes)
  g0[2] = (unsigned)(ga & 0xFFFFFFFFu);          // global_addr[31:0]
  g0[3] = (unsigned)((ga >> 32) & 0x01FFFFFFu)   // global_addr[56:32]
        | (2u << 30);                            // type = 2 ("image")
  tdm_g1_t g1;
  g1[0] = (int)((1u << 16)        // data_size = 2 bytes
              | (1u << 20)        // pad_enable
              | (4u << 22)        // pad_interval: 32 DWORDs
              | (3u << 25));      // pad_amount : 4 DWORDs
  const unsigned td0 = 1u << 20;  // tensor_dim0 (large: tile always in-bounds)
  const unsigned td1 = 1u << 20;  // tensor_dim1
  g1[1] = (int)((td0 & 0xFFFFu) << 16);                 // [63:48] = dim0 lo16
  g1[2] = (int)(((td0 >> 16) & 0xFFFFu) | ((td1 & 0xFFFFu) << 16));
  g1[3] = (int)(((td1 >> 16) & 0xFFFFu) | ((unsigned)LDSK << 16)); // tile_dim0
  g1[4] = (int)128u;                                    // tile_dim1=128, tile_dim2=0
  g1[5] = (int)stride_halves;                           // tensor_dim0_stride lo32
  g1[6] = 0;                                            // stride hi16 | dim1_stride
  g1[7] = 0;
  const tdm_g2_t zz = {0, 0, 0, 0};
  const tdm_g1_t z8 = {0, 0, 0, 0, 0, 0, 0, 0};
  __builtin_amdgcn_tensor_load_to_lds(g0, g1, zz, zz, z8, 0);
}

// ---- WMMA tile loads from LDS (f16, 16x16x32) -----------------------------
// A-tile 16x32 (MxK): lanes 0-15 row M=lane, K={0..7,16..23}; lanes 16-31
// row M=lane-16, K={8..15,24..31}.
__device__ inline v16h load_tile_A(const _Float16* __restrict__ base, int lane) {
  const int m  = lane & 15;
  const int hi = (lane >> 4) & 1;
  const _Float16* p = base + m * LDSS;
  v8h lo = *(const v8h*)(p + hi * 8);
  v8h hh = *(const v8h*)(p + 16 + hi * 8);
  v16h r;
#pragma unroll
  for (int j = 0; j < 8; ++j) { r[j] = lo[j]; r[j + 8] = hh[j]; }
  return r;
}

// B-tile 32x16 (KxN); LDS holds B transposed: base[n*LDSS + k] = B[k][n].
// lanes 0-15: col N=lane, K={0..7,16..23}; lanes 16-31: col N=lane-16, K={8..15,24..31}.
__device__ inline v16h load_tile_B(const _Float16* __restrict__ base, int lane) {
  const int n  = lane & 15;
  const int hi = (lane >> 4) & 1;
  const _Float16* p = base + n * LDSS;
  v8h lo = *(const v8h*)(p + hi * 8);
  v8h hh = *(const v8h*)(p + 16 + hi * 8);
  v16h r;
#pragma unroll
  for (int j = 0; j < 8; ++j) { r[j] = lo[j]; r[j + 8] = hh[j]; }
  return r;
}

__device__ inline v8f wmma16(v16h a, v16h b, v8f c) {
  return __builtin_amdgcn_wmma_f32_16x16x32_f16(false, a, false, b, (short)0, c,
                                                false, false);
}

// One K=32 sub-step: wave (wr in 0..3, wc in 0..1) owns a 32x64 patch of the
// 128x128 block; tiles preloaded into distinct registers, then 8 WMMAs.
__device__ inline void mma_step(const _Float16* __restrict__ As,
                                const _Float16* __restrict__ Bs,
                                int wr, int wc, int lane, v8f acc[2][4]) {
  const v16h a0 = load_tile_A(As + (wr * 32 + 0)  * LDSS, lane);
  const v16h a1 = load_tile_A(As + (wr * 32 + 16) * LDSS, lane);
  v16h b[4];
#pragma unroll
  for (int j = 0; j < 4; ++j)
    b[j] = load_tile_B(Bs + (wc * 64 + j * 16) * LDSS, lane);
#pragma unroll
  for (int j = 0; j < 4; ++j) {
    acc[0][j] = wmma16(a0, b[j], acc[0][j]);
    acc[1][j] = wmma16(a1, b[j], acc[1][j]);
  }
}

// ---- Manual staging (proj only: fuses f32 -> f16 conversion) --------------
__device__ inline void stage_f32(_Float16* __restrict__ dst,
                                 const float* __restrict__ src,
                                 size_t stride, int t) {
#pragma unroll
  for (int i = 0; i < 8; ++i) {                 // 2048 float4 groups
    int g = i * 256 + t;
    int r = g >> 4, c = (g & 15) * 4;
    float4 v = *(const float4*)(src + (size_t)r * stride + c);
    _Float16* d = dst + r * LDSS + c;
    d[0] = (_Float16)v.x; d[1] = (_Float16)v.y;
    d[2] = (_Float16)v.z; d[3] = (_Float16)v.w;
  }
}

// ---- Projection: Y = X(f32) @ W(f32)^T + b, f16 out -----------------------
template <int TRANSPOSED>
__global__ __launch_bounds__(256)
void proj_kernel(const float* __restrict__ X, const float* __restrict__ W,
                 const float* __restrict__ bias, _Float16* __restrict__ Y,
                 float scale) {
  __shared__ _Float16 Xs[2][128 * LDSS];
  __shared__ _Float16 Ws[2][128 * LDSS];
  const int t = threadIdx.x, lane = t & 31, wv = t >> 5;
  const int wr = wv & 3, wc = wv >> 2;
  const int row0 = blockIdx.x * 128;
  const int col0 = blockIdx.y * 128;
  const float* Xb = X + (size_t)row0 * DM;
  const float* Wb = W + (size_t)col0 * DM;

  v8f acc[2][4] = {};
  stage_f32(Xs[0], Xb, DM, t);
  stage_f32(Ws[0], Wb, DM, t);
  __syncthreads();
  int cur = 0;
  for (int d0 = 0; d0 < DM; d0 += LDSK) {
    if (d0 + LDSK < DM) {
      stage_f32(Xs[cur ^ 1], Xb + d0 + LDSK, DM, t);
      stage_f32(Ws[cur ^ 1], Wb + d0 + LDSK, DM, t);
    }
    mma_step(Xs[cur] + 0,  Ws[cur] + 0,  wr, wc, lane, acc);
    mma_step(Xs[cur] + 32, Ws[cur] + 32, wr, wc, lane, acc);
    __syncthreads();
    cur ^= 1;
  }

  const int n = lane & 15, mb = (lane >> 4) * 8;
#pragma unroll
  for (int j = 0; j < 4; ++j) {
    const int col = col0 + wc * 64 + j * 16 + n;
    const float bj = bias[col];
#pragma unroll
    for (int i = 0; i < 2; ++i) {
      const int rowb = row0 + wr * 32 + i * 16 + mb;
      if (TRANSPOSED) {
        v8h o;
#pragma unroll
        for (int r = 0; r < 8; ++r) o[r] = (_Float16)((acc[i][j][r] + bj) * scale);
        const int bidx = rowb >> 12;            // rowb / SEQ
        const int nn   = rowb & (SEQ - 1);
        *(v8h*)(Y + (size_t)bidx * DM * SEQ + (size_t)col * SEQ + nn) = o;
      } else {
#pragma unroll
        for (int r = 0; r < 8; ++r)
          Y[(size_t)(rowb + r) * DM + col] = (_Float16)((acc[i][j][r] + bj) * scale);
      }
    }
  }
}

// ---- Scores: S = Q @ K^T (softmax scale pre-folded into Q), f16 out -------
// TDM stages both operand tiles; wave 0 issues the DMA, everyone barriers.
__global__ __launch_bounds__(256)
void scores_kernel(const _Float16* __restrict__ Q, const _Float16* __restrict__ K,
                   _Float16* __restrict__ S) {
  __shared__ _Float16 Qs[2][128 * LDSS];
  __shared__ _Float16 Ks[2][128 * LDSS];
  const int t = threadIdx.x, lane = t & 31, wv = t >> 5;
  const int wr = wv & 3, wc = wv >> 2;
  const int row0 = blockIdx.x * 128;   // query rows
  const int col0 = blockIdx.y * 128;   // key columns
  const size_t z = blockIdx.z;
  const _Float16* Qb = Q + z * SEQ * DM + (size_t)row0 * DM;
  const _Float16* Kb = K + z * SEQ * DM + (size_t)col0 * DM;
  _Float16* Sb = S + z * SEQ * SEQ;

  v8f acc[2][4] = {};
  if (wv == 0) {
    tdm_load_2d(Qb, Qs[0], DM);
    tdm_load_2d(Kb, Ks[0], DM);
    __builtin_amdgcn_s_wait_tensorcnt((short)0);
  }
  __syncthreads();
  int cur = 0;
  for (int d0 = 0; d0 < DM; d0 += LDSK) {
    if (wv == 0 && d0 + LDSK < DM) {
      tdm_load_2d(Qb + d0 + LDSK, Qs[cur ^ 1], DM);
      tdm_load_2d(Kb + d0 + LDSK, Ks[cur ^ 1], DM);
    }
    mma_step(Qs[cur] + 0,  Ks[cur] + 0,  wr, wc, lane, acc);
    mma_step(Qs[cur] + 32, Ks[cur] + 32, wr, wc, lane, acc);
    if (wv == 0) __builtin_amdgcn_s_wait_tensorcnt((short)0);
    __syncthreads();
    cur ^= 1;
  }

  const int n = lane & 15, mb = (lane >> 4) * 8;
#pragma unroll
  for (int j = 0; j < 4; ++j)
#pragma unroll
    for (int i = 0; i < 2; ++i)
#pragma unroll
      for (int r = 0; r < 8; ++r) {
        const int row = row0 + wr * 32 + i * 16 + mb + r;
        const int col = col0 + wc * 64 + j * 16 + n;
        Sb[(size_t)row * SEQ + col] = (_Float16)acc[i][j][r];
      }
}

// ---- Softmax: one wave per 4096-wide row, in place, normalized ------------
__global__ __launch_bounds__(256)
void softmax_kernel(_Float16* __restrict__ S) {
  const int row  = blockIdx.x * 8 + (threadIdx.x >> 5);
  const int lane = threadIdx.x & 31;
  _Float16* p = S + (size_t)blockIdx.z * SEQ * SEQ + (size_t)row * SEQ;
  v8h v[16];
  float m = -1e30f;
#pragma unroll
  for (int j = 0; j < 16; ++j) {
    v[j] = *(const v8h*)(p + j * 256 + lane * 8);
#pragma unroll
    for (int e = 0; e < 8; ++e) m = fmaxf(m, (float)v[j][e]);
  }
#pragma unroll
  for (int off = 16; off > 0; off >>= 1) m = fmaxf(m, __shfl_xor(m, off, 32));
  float s = 0.0f;
#pragma unroll
  for (int j = 0; j < 16; ++j)
#pragma unroll
    for (int e = 0; e < 8; ++e) s += __expf((float)v[j][e] - m);
#pragma unroll
  for (int off = 16; off > 0; off >>= 1) s += __shfl_xor(s, off, 32);
  const float inv = 1.0f / s;
#pragma unroll
  for (int j = 0; j < 16; ++j) {
    v8h o;
#pragma unroll
    for (int e = 0; e < 8; ++e) o[e] = (_Float16)(__expf((float)v[j][e] - m) * inv);
    *(v8h*)(p + j * 256 + lane * 8) = o;
  }
}

// ---- PV: O = P @ V  (V supplied transposed per batch: VT[d][n]) -----------
__global__ __launch_bounds__(256)
void pv_kernel(const _Float16* __restrict__ P, const _Float16* __restrict__ VT,
               float* __restrict__ O) {
  __shared__ _Float16 Ps[2][128 * LDSS];
  __shared__ _Float16 Vs[2][128 * LDSS];
  const int t = threadIdx.x, lane = t & 31, wv = t >> 5;
  const int wr = wv & 3, wc = wv >> 2;
  const int row0 = blockIdx.x * 128;   // query rows
  const int col0 = blockIdx.y * 128;   // d columns
  const size_t z = blockIdx.z;
  const _Float16* Pb = P + z * SEQ * SEQ + (size_t)row0 * SEQ;
  const _Float16* Vb = VT + z * DM * SEQ + (size_t)col0 * SEQ;
  float* Ob = O + z * SEQ * DM;

  v8f acc[2][4] = {};
  if (wv == 0) {
    tdm_load_2d(Pb, Ps[0], SEQ);
    tdm_load_2d(Vb, Vs[0], SEQ);
    __builtin_amdgcn_s_wait_tensorcnt((short)0);
  }
  __syncthreads();
  int cur = 0;
  for (int k0 = 0; k0 < SEQ; k0 += LDSK) {
    if (wv == 0 && k0 + LDSK < SEQ) {
      tdm_load_2d(Pb + k0 + LDSK, Ps[cur ^ 1], SEQ);
      tdm_load_2d(Vb + k0 + LDSK, Vs[cur ^ 1], SEQ);
    }
    mma_step(Ps[cur] + 0,  Vs[cur] + 0,  wr, wc, lane, acc);
    mma_step(Ps[cur] + 32, Vs[cur] + 32, wr, wc, lane, acc);
    if (wv == 0) __builtin_amdgcn_s_wait_tensorcnt((short)0);
    __syncthreads();
    cur ^= 1;
  }

  const int n = lane & 15, mb = (lane >> 4) * 8;
#pragma unroll
  for (int j = 0; j < 4; ++j)
#pragma unroll
    for (int i = 0; i < 2; ++i)
#pragma unroll
      for (int r = 0; r < 8; ++r) {
        const int row = row0 + wr * 32 + i * 16 + mb + r;
        const int col = col0 + wc * 64 + j * 16 + n;
        Ob[(size_t)row * DM + col] = acc[i][j][r];
      }
}

// ---------------------------------------------------------------------------
extern "C" void kernel_launch(void* const* d_in, const int* in_sizes, int n_in,
                              void* d_out, int out_size, void* d_ws, size_t ws_size,
                              hipStream_t stream) {
  (void)in_sizes; (void)n_in; (void)out_size; (void)ws_size;
  const float* query = (const float*)d_in[0];
  const float* key_  = (const float*)d_in[1];
  const float* value = (const float*)d_in[2];
  const float* Wq    = (const float*)d_in[3];
  const float* bq    = (const float*)d_in[4];
  const float* Wk    = (const float*)d_in[5];
  const float* bk    = (const float*)d_in[6];
  const float* Wv    = (const float*)d_in[7];
  const float* bv    = (const float*)d_in[8];
  float* out = (float*)d_out;

  char* ws = (char*)d_ws;
  const size_t MB = (size_t)1 << 20;
  _Float16* Qh = (_Float16*)(ws + 0 * MB);    // [B*N, D] f16 (pre-scaled)
  _Float16* Kh = (_Float16*)(ws + 32 * MB);   // [B*N, D] f16
  _Float16* VT = (_Float16*)(ws + 64 * MB);   // [B][D][N] f16
  _Float16* S  = (_Float16*)(ws + 96 * MB);   // [2][N, N] f16 (batch pair)

  const float scale = 1.0f / sqrtf((float)DM);
  dim3 blk(256);
  dim3 gproj((NBATCH * SEQ) / 128, DM / 128);
  proj_kernel<0><<<gproj, blk, 0, stream>>>(query, Wq, bq, Qh, scale);
  proj_kernel<0><<<gproj, blk, 0, stream>>>(key_,  Wk, bk, Kh, 1.0f);
  proj_kernel<1><<<gproj, blk, 0, stream>>>(value, Wv, bv, VT, 1.0f);

  for (int b = 0; b < NBATCH; b += 2) {
    const _Float16* Qb = Qh + (size_t)b * SEQ * DM;
    const _Float16* Kb = Kh + (size_t)b * SEQ * DM;
    const _Float16* Vb = VT + (size_t)b * DM * SEQ;
    scores_kernel <<<dim3(SEQ / 128, SEQ / 128, 2), blk, 0, stream>>>(Qb, Kb, S);
    softmax_kernel<<<dim3(SEQ / 8, 1, 2),           blk, 0, stream>>>(S);
    pv_kernel     <<<dim3(SEQ / 128, DM / 128, 2),  blk, 0, stream>>>(S, Vb,
                                                  out + (size_t)b * SEQ * DM);
  }
}
